// NeuralCDE_10617159155881
// MI455X (gfx1250) — matrix operand order repaired and checked by reading, not compiled, not run on hardware
//
#include <hip/hip_runtime.h>
#include <hip/hip_bf16.h>

typedef __attribute__((ext_vector_type(16))) _Float16 v16h;
typedef __attribute__((ext_vector_type(8)))  _Float16 v8h;
typedef __attribute__((ext_vector_type(8)))  float    v8f;

#define B_   256
#define T_   200
#define C_   10
#define H_   128
#define HH_  128
#define O_   20
#define TM1  (T_ - 1)

// ---------------- workspace layout (bytes) ----------------
#define OFF_W1P   0         // 16384 halves  (32768 B)  WMMA-B fragments of W1
#define OFF_W2RP  32768     // 163840 halves (327680 B) fragments of per-channel W2r
#define OFF_B2R   360448    // 1280 f32                 b2 permuted
#define OFF_DT    365568    // 199 f32 (padded)
#define OFF_DXDT  366368    // 509440 f32

// B fragment layout this code uses for v_wmma_f32_16x16x32_f16 (wave32; contiguous k, same order as A):
// lane L<16  holds column N=L,    K = kb*32 + j      (j=0..15)
// lane L>=16 holds column N=L-16, K = kb*32 + 16 + j
// fragment stored contiguously: frag_base + lane*16 + j   (halves)

__global__ __launch_bounds__(256) void cde_prep(
    const float* __restrict__ coeffs, const float* __restrict__ times,
    const float* __restrict__ W1, const float* __restrict__ W2,
    const float* __restrict__ b2,
    _Float16* __restrict__ W1p, _Float16* __restrict__ W2rp,
    float* __restrict__ b2r, float* __restrict__ dtv, float* __restrict__ dXdt)
{
    const int gid = blockIdx.x * 256 + threadIdx.x;

    if (gid < 16384) {                       // ---- pack W1 [128,128] ----
        int p = gid;
        int j = p & 15, lane = (p >> 4) & 31, frag = p >> 9;
        int nb = frag & 7, kb = frag >> 3;
        int N = lane & 15;
        int K = kb * 32 + ((lane >> 4) << 4) + j;
        W1p[p] = (_Float16)W1[K * HH_ + nb * 16 + N];
    } else if (gid < 16384 + 163840) {       // ---- pack W2r [10][128][128] ----
        int q = gid - 16384;
        int j = q & 15, lane = (q >> 4) & 31, frag = q >> 9;
        int nb = frag & 7, kb = (frag >> 3) & 3, c = frag >> 5;
        int N = lane & 15;
        int K = kb * 32 + ((lane >> 4) << 4) + j;
        int h = nb * 16 + N;
        W2rp[q] = (_Float16)W2[K * (H_ * C_) + h * C_ + c];
    } else if (gid < 16384 + 163840 + 1280) { // ---- permute b2 ----
        int i = gid - (16384 + 163840);
        int c = i / H_, h = i % H_;
        b2r[i] = b2[h * C_ + c];
    } else if (gid < 16384 + 163840 + 1280 + TM1) { // ---- dt ----
        int t = gid - (16384 + 163840 + 1280);
        dtv[t] = times[t + 1] - times[t];
    } else if (gid < 16384 + 163840 + 1280 + TM1 + B_ * TM1 * C_) { // ---- dXdt ----
        int i = gid - (16384 + 163840 + 1280 + TM1);
        int c = i % C_;
        int t = (i / C_) % TM1;
        int b = i / (C_ * TM1);
        float dts = times[t + 1] - times[t];
        float x1 = coeffs[(b * T_ + t + 1) * C_ + c];
        float x0 = coeffs[(b * T_ + t) * C_ + c];
        dXdt[i] = (x1 - x0) / dts;
    }
}

// load A fragment (16x32 f16, M x K) for wave32 from a row-major [16][128] f16 LDS tile
__device__ __forceinline__ v16h load_a_frag(const _Float16* s, int lane, int kb) {
    const int row  = lane & 15;
    const int base = kb * 32 + ((lane >> 4) << 4);
    v8h lo = *(const v8h*)&s[row * H_ + base];
    v8h hi = *(const v8h*)&s[row * H_ + base + 8];
    return __builtin_shufflevector(lo, hi, 0, 1, 2, 3, 4, 5, 6, 7,
                                           8, 9, 10, 11, 12, 13, 14, 15);
}

// 512 threads = 16 wave32s. Wave w: wg = w>>3 (c/k split group), nb = w&7 (N-tile).
__global__ __launch_bounds__(512) void cde_scan(
    const float* __restrict__ coeffs, const float* __restrict__ W_init,
    const float* __restrict__ b_init, const float* __restrict__ b1,
    const float* __restrict__ W_out, const float* __restrict__ b_out,
    const _Float16* __restrict__ W1p, const _Float16* __restrict__ W2rp,
    const float* __restrict__ b2r, const float* __restrict__ dtv,
    const float* __restrict__ dXdt, float* __restrict__ out)
{
    __shared__ float    sZ[16 * H_];        // 8 KB   hidden state z (f32)
    __shared__ _Float16 sZin[16 * H_];      // 4 KB   stage input (f16)
    __shared__ _Float16 sH1[16 * H_];       // 4 KB   relu(z@W1+b1) (f16)
    __shared__ float    sK[4][16 * H_];     // 32 KB  RK4 stage outputs
    __shared__ float    sRed[16 * H_];      // 8 KB   cross-wave-group reduction
    __shared__ float    sDx[16 * C_];
    __shared__ float    sB1[H_];
    __shared__ float    sB2r[C_ * H_];

    const int tid  = threadIdx.x;
    const int lane = tid & 31;
    const int w    = tid >> 5;              // 0..15
    const int wg   = w >> 3;                // 0 or 1: split group
    const int nb   = w & 7;                 // N-tile 0..7
    const int blk  = blockIdx.x;            // batch tile 0..15
    const int n16  = lane & 15;
    const int M0   = (lane >> 4) << 3;      // 0 or 8
    const int colN = nb * 16 + n16;

    if (tid < H_) sB1[tid] = b1[tid];
    for (int i = tid; i < C_ * H_; i += 512) sB2r[i] = b2r[i];

    // W1 fragments for this wave's (kb, nb): kb = 2*wg, 2*wg+1 — pinned in VGPRs
    const int kb0 = wg * 2;
    const v16h w1f0 = *(const v16h*)(W1p + ((((kb0 + 0) << 3) + nb) * 32 + lane) * 16);
    const v16h w1f1 = *(const v16h*)(W1p + ((((kb0 + 1) << 3) + nb) * 32 + lane) * 16);

    // ---- z0 = coeffs[:,0] @ W_init + b_init ----
    for (int i = tid; i < 16 * H_; i += 512) {
        int m = i >> 7, hh = i & (H_ - 1);
        int b = blk * 16 + m;
        float acc = b_init[hh];
#pragma unroll
        for (int c = 0; c < C_; ++c)
            acc += coeffs[(b * T_) * C_ + c] * W_init[c * H_ + hh];
        sZ[i] = acc;
    }
    __syncthreads();

    for (int t = 0; t < TM1; ++t) {
        const float h = dtv[t];
        for (int i = tid; i < 16 * C_; i += 512) {
            int m = i / C_, c = i % C_;
            sDx[i] = dXdt[((blk * 16 + m) * TM1 + t) * C_ + c];
        }
        __syncthreads();

#pragma unroll 1
        for (int stage = 0; stage < 4; ++stage) {
            // ---- build stage input zin = z + alpha * k_prev (f16) ----
            const float alpha = (stage == 0) ? 0.0f : (stage == 3 ? h : 0.5f * h);
            for (int i = tid; i < 16 * H_; i += 512) {
                float zi = sZ[i];
                if (stage > 0) zi += alpha * sK[stage - 1][i];
                sZin[i] = (_Float16)zi;
            }
            __syncthreads();

            // ---- stage A: h1 = relu(zin @ W1 + b1); K split across wave groups ----
            v8f acc = {};
            {
                v16h a0 = load_a_frag(sZin, lane, kb0 + 0);
                acc = __builtin_amdgcn_wmma_f32_16x16x32_f16(
                    false, a0, false, w1f0, (short)0, acc, false, false);
                v16h a1 = load_a_frag(sZin, lane, kb0 + 1);
                acc = __builtin_amdgcn_wmma_f32_16x16x32_f16(
                    false, a1, false, w1f1, (short)0, acc, false, false);
            }
            if (wg == 0) {
#pragma unroll
                for (int r = 0; r < 8; ++r)
                    sRed[(M0 + r) * H_ + colN] = acc[r];
            }
            __syncthreads();
            if (wg == 1) {
                const float bias = sB1[colN];
#pragma unroll
                for (int r = 0; r < 8; ++r) {
                    float v = acc[r] + sRed[(M0 + r) * H_ + colN] + bias;
                    v = fmaxf(v, 0.0f);
                    sH1[(M0 + r) * H_ + colN] = (_Float16)v;
                }
            }
            __syncthreads();

            // ---- stage B: fout = sum_c tanh(h1 @ W2r[c] + b2r[c]) * dxdt[:,c]
            //      c split across wave groups: wg 0 -> c=0..4, wg 1 -> c=5..9 ----
            v8f fout = {};
#pragma unroll 1
            for (int ci = 0; ci < 5; ++ci) {
                const int cc = wg * 5 + ci;
                if (ci < 4)
                    __builtin_prefetch(W2rp + (size_t)(cc + 1) * 16384, 0, 1);
                v8f g = {};
#pragma unroll
                for (int kb = 0; kb < 4; ++kb) {
                    v16h a = load_a_frag(sH1, lane, kb);
                    v16h bm = *(const v16h*)(W2rp +
                        ((((cc << 2) + kb) << 3) + nb) * 512 + lane * 16);
                    g = __builtin_amdgcn_wmma_f32_16x16x32_f16(
                        false, a, false, bm, (short)0, g, false, false);
                }
                const float bias = sB2r[cc * H_ + colN];
#pragma unroll
                for (int r = 0; r < 8; ++r) {
                    float gg = tanhf(g[r] + bias);
                    fout[r] += gg * sDx[(M0 + r) * C_ + cc];
                }
            }
            if (wg == 0) {
#pragma unroll
                for (int r = 0; r < 8; ++r)
                    sRed[(M0 + r) * H_ + colN] = fout[r];
            }
            __syncthreads();
            if (wg == 1) {
#pragma unroll
                for (int r = 0; r < 8; ++r)
                    sK[stage][(M0 + r) * H_ + colN] =
                        fout[r] + sRed[(M0 + r) * H_ + colN];
            }
            __syncthreads();
        }

        // ---- RK4 combine ----
        const float h6 = h * (1.0f / 6.0f);
        for (int i = tid; i < 16 * H_; i += 512)
            sZ[i] += h6 * (sK[0][i] + 2.0f * sK[1][i] + 2.0f * sK[2][i] + sK[3][i]);
        __syncthreads();
    }

    // ---- head: out = zT @ W_out + b_out ----
    for (int idx = tid; idx < 16 * O_; idx += 512) {
        int m = idx / O_, o = idx % O_;
        float acc = b_out[o];
        for (int hh = 0; hh < H_; ++hh)
            acc += sZ[m * H_ + hh] * W_out[hh * O_ + o];
        out[(blk * 16 + m) * O_ + o] = acc;
    }
}

extern "C" void kernel_launch(void* const* d_in, const int* in_sizes, int n_in,
                              void* d_out, int out_size, void* d_ws, size_t ws_size,
                              hipStream_t stream) {
    const float* coeffs = (const float*)d_in[0];
    const float* times  = (const float*)d_in[1];
    const float* W_init = (const float*)d_in[2];
    const float* b_init = (const float*)d_in[3];
    const float* W1     = (const float*)d_in[4];
    const float* b1     = (const float*)d_in[5];
    const float* W2     = (const float*)d_in[6];
    const float* b2     = (const float*)d_in[7];
    const float* W_out  = (const float*)d_in[8];
    const float* b_out  = (const float*)d_in[9];
    float* out = (float*)d_out;

    char* ws = (char*)d_ws;
    _Float16* W1p  = (_Float16*)(ws + OFF_W1P);
    _Float16* W2rp = (_Float16*)(ws + OFF_W2RP);
    float* b2r  = (float*)(ws + OFF_B2R);
    float* dtv  = (float*)(ws + OFF_DT);
    float* dXdt = (float*)(ws + OFF_DXDT);

    const int total = 16384 + 163840 + 1280 + TM1 + B_ * TM1 * C_;
    const int prep_blocks = (total + 255) / 256;
    cde_prep<<<prep_blocks, 256, 0, stream>>>(coeffs, times, W1, W2, b2,
                                              W1p, W2rp, b2r, dtv, dXdt);
    cde_scan<<<B_ / 16, 512, 0, stream>>>(coeffs, W_init, b_init, b1,
                                          W_out, b_out, W1p, W2rp, b2r,
                                          dtv, dXdt, out);
}
